// FuncConv_36455682409148
// MI455X (gfx1250) — matrix operand order repaired and checked by reading, not compiled, not run on hardware
//
#include <hip/hip_runtime.h>

typedef __attribute__((ext_vector_type(16))) __bf16 v16bf;
typedef __attribute__((ext_vector_type(8)))  float  v8f;

#define LLD __device__ __forceinline__

#define TPB   128
#define WAVES 4
#define HSTR  132   // f32 row stride (padding: bank stride 4)
#define YSTR  68    // bf16 row stride (padding)

// ---- CDNA5 async global->LDS path (guarded so compilation never breaks) ----
#ifdef __has_builtin
#if __has_builtin(__builtin_amdgcn_global_load_async_to_lds_b128)
#define USE_ASYNC_LDS 1
#endif
#if __has_builtin(__builtin_amdgcn_s_wait_asynccnt)
#define HAVE_WAIT_ASYNC_BUILTIN 1
#endif
#endif

#if defined(USE_ASYNC_LDS)
typedef int v4i __attribute__((vector_size(4 * sizeof(int))));
typedef __attribute__((address_space(1))) v4i gas_v4i;
typedef __attribute__((address_space(3))) v4i lds_v4i;
LLD void async_copy_b128(const float* g, float* l) {
  __builtin_amdgcn_global_load_async_to_lds_b128((gas_v4i*)g, (lds_v4i*)l, 0, 0);
}
LLD void wait_async0() {
#if defined(HAVE_WAIT_ASYNC_BUILTIN)
  __builtin_amdgcn_s_wait_asynccnt(0);
#else
  asm volatile("s_wait_asynccnt 0x0" ::: "memory");
#endif
  __builtin_amdgcn_wave_barrier();
}
#endif

// Wave-local LDS sync: all producers/consumers are lanes of the same wave,
// DS ops are in-order per wave -> waiting DScnt==0 makes writes visible.
LLD void wave_lds_sync() {
  __builtin_amdgcn_wave_barrier();
  asm volatile("s_wait_dscnt 0x0" ::: "memory");
  __builtin_amdgcn_wave_barrier();
}

LLD float lrelu(float x) { return fmaxf(x, 0.01f * x); }

LLD v8f splat8(float x) { v8f c = {x, x, x, x, x, x, x, x}; return c; }

// B operand: pre-swizzled blob, lane-contiguous 16 bf16 (32B -> b128 pair).
LLD v16bf loadB(const __bf16* blob, int kt, int NT, int nt, int lane) {
  const __bf16* p = blob + (((kt * NT + nt) * 32 + lane) << 4);
  v16bf b;
#pragma unroll
  for (int e = 0; e < 16; ++e) b[e] = p[e];
  return b;
}

// A operand (16-bit A 16x32 layout): lane half=lane/16, row M=lane%16.
// e<8 : K = half*8 + e ; e>=8 : K = 16 + half*8 + (e-8)
LLD v16bf loadA_f32(const float* row, int kt, int half, float scale) {
  v16bf a;
#pragma unroll
  for (int e = 0; e < 16; ++e) {
    int K = kt * 32 + ((e < 8) ? (half * 8 + e) : (8 + half * 8 + e));
    a[e] = (__bf16)(row[K] * scale);
  }
  return a;
}

LLD v16bf loadA_bf16(const __bf16* row, int kt, int half) {
  v16bf a;
#pragma unroll
  for (int e = 0; e < 16; ++e) {
    int K = kt * 32 + ((e < 8) ? (half * 8 + e) : (8 + half * 8 + e));
    a[e] = row[K];
  }
  return a;
}

// Hidden layer: D = lrelu(A @ W^T + b), result stored bf16 in row-padded LDS.
template <int KT, int NT>
LLD void hidden_layer(const v16bf* A, const __bf16* blob, const float* bias,
                      __bf16* yb, int lane) {
  const int half = lane >> 4, mcol = lane & 15;
#pragma unroll
  for (int nt = 0; nt < NT; ++nt) {
    v8f c = splat8(bias[nt * 16 + mcol]);
#pragma unroll
    for (int kt = 0; kt < KT; ++kt) {
      v16bf B = loadB(blob, kt, NT, nt, lane);
      c = __builtin_amdgcn_wmma_f32_16x16x32_bf16(false, A[kt], false, B,
                                                  (short)0, c, false, false);
    }
#pragma unroll
    for (int v = 0; v < 8; ++v)
      yb[(v + 8 * half) * YSTR + nt * 16 + mcol] = (__bf16)lrelu(c[v]);
  }
}

// Stage weights (row-major [Nout][Kin] f32) into bf16 B-operand blob order:
// blob[((kt*NT+nt)*32 + lane)*16 + e] = w[nt*16 + lane%16][kt*32 + (e<8 ? (lane/16)*8 + e : 16 + (lane/16)*8 + (e-8))]
template <int KT, int NT>
LLD void stage_blob(const float* w, __bf16* blob, int Kin) {
  const int total = KT * NT * 512;
  for (int i = threadIdx.x; i < total; i += blockDim.x) {
    int e  = i & 15;
    int ln = (i >> 4) & 31;
    int t  = i >> 9;
    int nt = t % NT, kt = t / NT;
    int n = nt * 16 + (ln & 15);
    int k = kt * 32 + ((e < 8) ? (((ln >> 4) << 3) + e) : (8 + ((ln >> 4) << 3) + e));
    blob[i] = (__bf16)w[n * Kin + k];
  }
}

LLD void stage_bias(const float* b, float* bs, int n) {
  for (int i = threadIdx.x; i < n; i += blockDim.x) bs[i] = b[i];
}

// ---------------- Edge pass: m = (r ? mlp_inv(feat[src]) : feat[src]); atomic mean-scatter
__global__ __launch_bounds__(TPB) void edge_kernel(
    const float* __restrict__ feat, const int* __restrict__ src,
    const int* __restrict__ dst, const int* __restrict__ r,
    const float* iw1, const float* ib1, const float* iw2, const float* ib2,
    const float* iw3, const float* ib3,
    float* __restrict__ neigh, float* __restrict__ deg, int E, int ntiles) {
  __shared__ __bf16 wb1[4 * 4 * 512];
  __shared__ __bf16 wb2[2 * 4 * 512];
  __shared__ __bf16 wb3[2 * 8 * 512];
  __shared__ float  bs1[64], bs2[64], bs3[128];
  __shared__ float  hbuf[WAVES][16 * HSTR];
  __shared__ __bf16 yAb[WAVES][16 * YSTR];
  __shared__ __bf16 yBb[WAVES][16 * YSTR];

  stage_blob<4, 4>(iw1, wb1, 128);
  stage_blob<2, 4>(iw2, wb2, 64);
  stage_blob<2, 8>(iw3, wb3, 64);
  stage_bias(ib1, bs1, 64); stage_bias(ib2, bs2, 64); stage_bias(ib3, bs3, 128);
  __syncthreads();

  const int lane = threadIdx.x & 31;
  const int w    = threadIdx.x >> 5;
  const int half = lane >> 4, mcol = lane & 15;
  const int nwaves = (gridDim.x * TPB) >> 5;
  const int wid    = (blockIdx.x * TPB + threadIdx.x) >> 5;

  float*  hb = hbuf[w];
  __bf16* ya = yAb[w];
  __bf16* yb = yBb[w];

  for (int tile = wid; tile < ntiles; tile += nwaves) {
    const int base = tile * 16;
    // gather 16 feature rows -> LDS (128 f32 per row)
#if defined(USE_ASYNC_LDS)
    for (int i = lane; i < 512; i += 32) {
      int row = i >> 5;
      int eidx = base + row; if (eidx >= E) eidx = E - 1;
      int s = src[eidx];
      async_copy_b128(feat + (size_t)s * 128 + (size_t)(i & 31) * 4,
                      &hb[row * HSTR + (i & 31) * 4]);
    }
#else
    for (int i = lane; i < 512; i += 32) {
      int row = i >> 5;
      int eidx = base + row; if (eidx >= E) eidx = E - 1;
      int s = src[eidx];
      const float4 v4 = *(const float4*)(feat + (size_t)s * 128 + (size_t)(i & 31) * 4);
      *(float4*)&hb[row * HSTR + (i & 31) * 4] = v4;
    }
#endif
    int dstv[8], rvv[8];
#pragma unroll
    for (int v = 0; v < 8; ++v) {
      int m = v + 8 * half;
      int eidx = base + m;
      bool ok = eidx < E; if (!ok) eidx = E - 1;
      dstv[v] = ok ? dst[eidx] : -1;
      rvv[v]  = r[eidx];
    }
#if defined(USE_ASYNC_LDS)
    wait_async0();
#endif
    wave_lds_sync();

    v16bf A1[4];
#pragma unroll
    for (int kt = 0; kt < 4; ++kt)
      A1[kt] = loadA_f32(hb + mcol * HSTR, kt, half, 1.0f);
    hidden_layer<4, 4>(A1, wb1, bs1, ya, lane);
    wave_lds_sync();

    v16bf A2[2];
#pragma unroll
    for (int kt = 0; kt < 2; ++kt) A2[kt] = loadA_bf16(ya + mcol * YSTR, kt, half);
    hidden_layer<2, 4>(A2, wb2, bs2, yb, lane);
    wave_lds_sync();

    v16bf A3[2];
#pragma unroll
    for (int kt = 0; kt < 2; ++kt) A3[kt] = loadA_bf16(yb + mcol * YSTR, kt, half);
#pragma unroll
    for (int nt = 0; nt < 8; ++nt) {
      v8f c = splat8(bs3[nt * 16 + mcol]);
#pragma unroll
      for (int kt = 0; kt < 2; ++kt) {
        v16bf B = loadB(wb3, kt, 8, nt, lane);
        c = __builtin_amdgcn_wmma_f32_16x16x32_bf16(false, A3[kt], false, B,
                                                    (short)0, c, false, false);
      }
      const int col = nt * 16 + mcol;
#pragma unroll
      for (int v = 0; v < 8; ++v) {
        if (dstv[v] >= 0) {
          float hval = hb[(v + 8 * half) * HSTR + col];
          float mval = rvv[v] ? c[v] : hval;
          atomicAdd(&neigh[(size_t)dstv[v] * 128 + col], mval);
        }
      }
    }
    if (lane < 16) {
      int eidx = base + lane;
      if (eidx < E) atomicAdd(&deg[dst[eidx]], 1.0f);
    }
  }
}

// ---------------- Node pass: res = mlp_and(neigh/deg); out = inv ? mlp_inv(res) : res
__global__ __launch_bounds__(TPB) void node_kernel(
    const float* __restrict__ neigh, const float* __restrict__ deg,
    const int* __restrict__ inv,
    const float* aw1, const float* ab1, const float* aw2, const float* ab2,
    const float* aw3, const float* ab3,
    const float* iw1, const float* ib1, const float* iw2, const float* ib2,
    const float* iw3, const float* ib3,
    float* __restrict__ out, int N, int ntiles) {
  __shared__ __bf16 awb1[8192], awb2[4096], awb3[8192];
  __shared__ __bf16 iwb1[8192], iwb2[4096], iwb3[8192];
  __shared__ float  abs1[64], abs2[64], abs3[128];
  __shared__ float  ibs1[64], ibs2[64], ibs3[128];
  __shared__ float  xbuf[WAVES][16 * HSTR];
  __shared__ __bf16 yAb[WAVES][16 * YSTR];
  __shared__ __bf16 yBb[WAVES][16 * YSTR];
  __shared__ float  sclb[WAVES][16];

  stage_blob<4, 4>(aw1, awb1, 128);
  stage_blob<2, 4>(aw2, awb2, 64);
  stage_blob<2, 8>(aw3, awb3, 64);
  stage_blob<4, 4>(iw1, iwb1, 128);
  stage_blob<2, 4>(iw2, iwb2, 64);
  stage_blob<2, 8>(iw3, iwb3, 64);
  stage_bias(ab1, abs1, 64); stage_bias(ab2, abs2, 64); stage_bias(ab3, abs3, 128);
  stage_bias(ib1, ibs1, 64); stage_bias(ib2, ibs2, 64); stage_bias(ib3, ibs3, 128);
  __syncthreads();

  const int lane = threadIdx.x & 31;
  const int w    = threadIdx.x >> 5;
  const int half = lane >> 4, mcol = lane & 15;
  const int nwaves = (gridDim.x * TPB) >> 5;
  const int wid    = (blockIdx.x * TPB + threadIdx.x) >> 5;

  float*  xb  = xbuf[w];
  __bf16* ya  = yAb[w];
  __bf16* yb  = yBb[w];
  float*  scl = sclb[w];

  for (int tile = wid; tile < ntiles; tile += nwaves) {
    const int base = tile * 16;
    // load raw neighborhood rows -> LDS; mean scale applied at A-operand build
#if defined(USE_ASYNC_LDS)
    for (int i = lane; i < 512; i += 32) {
      int row = i >> 5;
      int node = base + row; if (node >= N) node = N - 1;
      async_copy_b128(neigh + (size_t)node * 128 + (size_t)(i & 31) * 4,
                      &xb[row * HSTR + (i & 31) * 4]);
    }
#else
    for (int i = lane; i < 512; i += 32) {
      int row = i >> 5;
      int node = base + row; if (node >= N) node = N - 1;
      const float4 v4 = *(const float4*)(neigh + (size_t)node * 128 + (size_t)(i & 31) * 4);
      *(float4*)&xb[row * HSTR + (i & 31) * 4] = v4;
    }
#endif
    if (lane < 16) {
      int node = base + lane; if (node >= N) node = N - 1;
      scl[lane] = 1.0f / fmaxf(deg[node], 1.0f);
    }
    int invv[8];
#pragma unroll
    for (int v = 0; v < 8; ++v) {
      int node = base + v + 8 * half; if (node >= N) node = N - 1;
      invv[v] = inv[node];
    }
#if defined(USE_ASYNC_LDS)
    wait_async0();
#endif
    wave_lds_sync();

    // ---- mlp_and ----
    const float sc = scl[mcol];
    v16bf A1[4];
#pragma unroll
    for (int kt = 0; kt < 4; ++kt)
      A1[kt] = loadA_f32(xb + mcol * HSTR, kt, half, sc);
    hidden_layer<4, 4>(A1, awb1, abs1, ya, lane);
    wave_lds_sync();
    v16bf A2[2];
#pragma unroll
    for (int kt = 0; kt < 2; ++kt) A2[kt] = loadA_bf16(ya + mcol * YSTR, kt, half);
    hidden_layer<2, 4>(A2, awb2, abs2, yb, lane);
    wave_lds_sync();
    v16bf A3[2];
#pragma unroll
    for (int kt = 0; kt < 2; ++kt) A3[kt] = loadA_bf16(yb + mcol * YSTR, kt, half);
#pragma unroll
    for (int nt = 0; nt < 8; ++nt) {
      v8f c = splat8(abs3[nt * 16 + mcol]);
#pragma unroll
      for (int kt = 0; kt < 2; ++kt) {
        v16bf B = loadB(awb3, kt, 8, nt, lane);
        c = __builtin_amdgcn_wmma_f32_16x16x32_bf16(false, A3[kt], false, B,
                                                    (short)0, c, false, false);
      }
#pragma unroll
      for (int v = 0; v < 8; ++v)   // res -> xbuf (x fully consumed in layer 1)
        xb[(v + 8 * half) * HSTR + nt * 16 + mcol] = c[v];
    }
    wave_lds_sync();

    // ---- mlp_inv(res), select, store ----
#pragma unroll
    for (int kt = 0; kt < 4; ++kt)
      A1[kt] = loadA_f32(xb + mcol * HSTR, kt, half, 1.0f);
    hidden_layer<4, 4>(A1, iwb1, ibs1, ya, lane);
    wave_lds_sync();
#pragma unroll
    for (int kt = 0; kt < 2; ++kt) A2[kt] = loadA_bf16(ya + mcol * YSTR, kt, half);
    hidden_layer<2, 4>(A2, iwb2, ibs2, yb, lane);
    wave_lds_sync();
#pragma unroll
    for (int kt = 0; kt < 2; ++kt) A3[kt] = loadA_bf16(yb + mcol * YSTR, kt, half);
#pragma unroll
    for (int nt = 0; nt < 8; ++nt) {
      v8f c = splat8(ibs3[nt * 16 + mcol]);
#pragma unroll
      for (int kt = 0; kt < 2; ++kt) {
        v16bf B = loadB(iwb3, kt, 8, nt, lane);
        c = __builtin_amdgcn_wmma_f32_16x16x32_bf16(false, A3[kt], false, B,
                                                    (short)0, c, false, false);
      }
      const int col = nt * 16 + mcol;
#pragma unroll
      for (int v = 0; v < 8; ++v) {
        int node = base + v + 8 * half;
        if (node < N) {
          float res = xb[(v + 8 * half) * HSTR + col];
          out[(size_t)node * 128 + col] = invv[v] ? c[v] : res;
        }
      }
    }
  }
}

__global__ void zero_kernel(float4* __restrict__ p4, size_t n4,
                            float* __restrict__ tail, size_t ntail) {
  size_t i = (size_t)blockIdx.x * blockDim.x + threadIdx.x;
  size_t stride = (size_t)gridDim.x * blockDim.x;
  float4 z = make_float4(0.f, 0.f, 0.f, 0.f);
  for (size_t j = i; j < n4; j += stride) p4[j] = z;
  for (size_t j = i; j < ntail; j += stride) tail[j] = 0.0f;
}

extern "C" void kernel_launch(void* const* d_in, const int* in_sizes, int n_in,
                              void* d_out, int out_size, void* d_ws, size_t ws_size,
                              hipStream_t stream) {
  const float* feat = (const float*)d_in[0];
  const int*   src  = (const int*)d_in[1];
  const int*   dst  = (const int*)d_in[2];
  const int*   r    = (const int*)d_in[3];
  const int*   inv  = (const int*)d_in[4];
  const float* iw1 = (const float*)d_in[5];
  const float* ib1 = (const float*)d_in[6];
  const float* iw2 = (const float*)d_in[7];
  const float* ib2 = (const float*)d_in[8];
  const float* iw3 = (const float*)d_in[9];
  const float* ib3 = (const float*)d_in[10];
  const float* aw1 = (const float*)d_in[11];
  const float* ab1 = (const float*)d_in[12];
  const float* aw2 = (const float*)d_in[13];
  const float* ab2 = (const float*)d_in[14];
  const float* aw3 = (const float*)d_in[15];
  const float* ab3 = (const float*)d_in[16];
  float* out = (float*)d_out;

  const int N = in_sizes[0] / 128;
  const int E = in_sizes[1];

  float* neigh = (float*)d_ws;
  float* deg   = neigh + (size_t)N * 128;

  // zero accumulators every call (deterministic; atomics accumulate)
  zero_kernel<<<2048, 256, 0, stream>>>((float4*)neigh, (size_t)N * 32,
                                        deg, (size_t)N);

  const int etiles = (E + 15) / 16;
  int eblocks = (etiles + WAVES - 1) / WAVES;
  if (eblocks > 2048) eblocks = 2048;
  edge_kernel<<<eblocks, TPB, 0, stream>>>(feat, src, dst, r,
                                           iw1, ib1, iw2, ib2, iw3, ib3,
                                           neigh, deg, E, etiles);

  const int ntiles = (N + 15) / 16;
  int nblocks = (ntiles + WAVES - 1) / WAVES;
  if (nblocks > 1024) nblocks = 1024;
  node_kernel<<<nblocks, TPB, 0, stream>>>(neigh, deg, inv,
                                           aw1, ab1, aw2, ab2, aw3, ab3,
                                           iw1, ib1, iw2, ib2, iw3, ib3,
                                           out, N, ntiles);
}
